// SelfAttention_8057358648211
// MI455X (gfx1250) — matrix operand order repaired and checked by reading, not compiled, not run on hardware
//
#include <hip/hip_runtime.h>
#include <hip/hip_bf16.h>

typedef __attribute__((ext_vector_type(16))) __bf16 v16bf;
typedef __attribute__((ext_vector_type(8)))  float  v8f;

#define BATCH 2
#define CCH   128
#define NSPAT 4096
#define NH    4
#define HD    32
#define NGRP  16
#define CPG   8   // channels per group

// ---------------------------------------------------------------------------
// Kernel 1: GroupNorm statistics: mean/rstd per (b, group) over CPG*NSPAT elems
// ---------------------------------------------------------------------------
__global__ void __launch_bounds__(256) gn_stats(const float* __restrict__ x,
                                                float* __restrict__ stats) {
    int bg = blockIdx.x;              // b*NGRP + g
    int b = bg / NGRP, g = bg % NGRP;
    const float* xp = x + ((size_t)(b * CCH + g * CPG)) * NSPAT;
    float s = 0.f, ss = 0.f;
    for (int i = threadIdx.x; i < CPG * NSPAT; i += 256) {
        float v = xp[i];
        s += v; ss += v * v;
    }
    __shared__ float ls[256], lss[256];
    ls[threadIdx.x] = s; lss[threadIdx.x] = ss;
    __syncthreads();
    for (int st = 128; st > 0; st >>= 1) {
        if (threadIdx.x < st) {
            ls[threadIdx.x]  += ls[threadIdx.x + st];
            lss[threadIdx.x] += lss[threadIdx.x + st];
        }
        __syncthreads();
    }
    if (threadIdx.x == 0) {
        const float inv = 1.0f / (float)(CPG * NSPAT);
        float mean = ls[0] * inv;
        float var  = lss[0] * inv - mean * mean;
        stats[2 * bg]     = mean;
        stats[2 * bg + 1] = rsqrtf(var + 1e-5f);
    }
}

// ---------------------------------------------------------------------------
// Kernel 2: apply GroupNorm (gamma/beta), emit bf16 normalized tensor [B][C][N]
// ---------------------------------------------------------------------------
__global__ void __launch_bounds__(256) gn_apply(const float* __restrict__ x,
                                                const float* __restrict__ gamma,
                                                const float* __restrict__ beta,
                                                const float* __restrict__ stats,
                                                __bf16* __restrict__ normbf) {
    int i = blockIdx.x * 256 + threadIdx.x;      // over B*C*N = 1,048,576
    int c = (i / NSPAT) % CCH;
    int b = i / (CCH * NSPAT);
    int g = c >> 3;
    float mean = stats[2 * (b * NGRP + g)];
    float rstd = stats[2 * (b * NGRP + g) + 1];
    float v = (x[i] - mean) * rstd * gamma[c] + beta[c];
    normbf[i] = (__bf16)v;
}

// ---------------------------------------------------------------------------
// Kernel 3: QKV projection GEMM via WMMA bf16.
//   out[o,n] = sum_c wqkv[o,c]*norm[c,n], o in [0,384), n in [0,4096)
//   Block: 128 threads (4 waves), tile 16(M) x 64(N), K=128 in 4 WMMA steps.
//   Epilogue scatters Q (scaled), K as [bh][n][hd]; V transposed [bh][hd][n].
// ---------------------------------------------------------------------------
__global__ void __launch_bounds__(128) qkv_gemm(const float*  __restrict__ wqkv,
                                                const __bf16* __restrict__ normbf,
                                                __bf16* __restrict__ qbf,
                                                __bf16* __restrict__ kbf,
                                                __bf16* __restrict__ vbf) {
    __shared__ __align__(16) __bf16 lds_a[16 * 128];    // weights tile [m][c]
    __shared__ __align__(16) __bf16 lds_b[64 * 128];    // input tile transposed [n][c]
    int mt = blockIdx.x;                  // 0..23
    int n0 = blockIdx.y * 64;             // 0..4032
    int b  = blockIdx.z;
    int tid = threadIdx.x;

    for (int i = tid; i < 16 * 128; i += 128) {
        int r = i >> 7, c0 = i & 127;
        lds_a[i] = (__bf16)wqkv[(mt * 16 + r) * 128 + c0];
    }
    for (int i = tid; i < 64 * 128; i += 128) {
        int c0 = i >> 6, j = i & 63;      // coalesced in n for global read
        lds_b[j * 128 + c0] = normbf[((size_t)(b * CCH + c0)) * NSPAT + n0 + j];
    }
    __syncthreads();

    int lane = tid & 31, wave = tid >> 5;
    int half = lane >> 4, l16 = lane & 15;
    v8f acc = {};
#pragma unroll
    for (int ks = 0; ks < 4; ++ks) {
        int kk0 = ks * 32;
        v16bf a, bb;
#pragma unroll
        for (int j = 0; j < 8; ++j) {
            // A fragment (16x32 M x K): contiguous k = 16*half + s, lane half selects K block
            int ka = half * 16 + 2 * j;
            a[2 * j]     = lds_a[l16 * 128 + kk0 + ka];
            a[2 * j + 1] = lds_a[l16 * 128 + kk0 + ka + 1];
            // B fragment (32x16 K x N): lanes 0-15 K=0..15, lanes 16-31 K=16..31
            int kb = half * 16 + 2 * j;
            bb[2 * j]     = lds_b[(wave * 16 + l16) * 128 + kk0 + kb];
            bb[2 * j + 1] = lds_b[(wave * 16 + l16) * 128 + kk0 + kb + 1];
        }
        acc = __builtin_amdgcn_wmma_f32_16x16x32_bf16(false, a, false, bb,
                                                      (short)0, acc, false, false);
    }

    const float scale = 0.088388347648318447f;    // 1/sqrt(128)
#pragma unroll
    for (int r = 0; r < 8; ++r) {
        int mloc = r + half * 8;                  // C/D layout row
        int o = mt * 16 + mloc;                   // 0..383
        int n = n0 + wave * 16 + l16;
        float val = acc[r];
        int region = o >> 7;                      // 0=q 1=k 2=v
        int c128 = o & 127;
        int hh = c128 >> 5, cc = c128 & 31;
        size_t bh = (size_t)(b * NH + hh);
        if (region == 0)      qbf[(bh * NSPAT + n) * HD + cc] = (__bf16)(val * scale);
        else if (region == 1) kbf[(bh * NSPAT + n) * HD + cc] = (__bf16)val;
        else                  vbf[(bh * HD + cc) * NSPAT + n] = (__bf16)val;
    }
}

// ---------------------------------------------------------------------------
// Kernel 4: flash attention, bf16 WMMA with online softmax.
//   grid: (N/128, NH, B); block 128 threads = 4 waves; wave owns 32 query rows
//   (two 16-row A-fragments sharing each staged K/V chunk).
//   Per 64-key chunk: stage K (64x32) + V^T (32x64) in LDS once per block,
//   then per wave: 2 x [4 WMMA S, online softmax, 4 WMMA PV].
// ---------------------------------------------------------------------------
__global__ void __launch_bounds__(128) flash_attn(const __bf16* __restrict__ qbf,
                                                  const __bf16* __restrict__ kbf,
                                                  const __bf16* __restrict__ vbf,
                                                  float* __restrict__ ao) {
    int b = blockIdx.z, h = blockIdx.y;
    int tid = threadIdx.x, lane = tid & 31, wave = tid >> 5;
    int half = lane >> 4, l16 = lane & 15;
    size_t bh = (size_t)(b * NH + h);
    int q0 = blockIdx.x * 128 + wave * 32;

    __shared__ __align__(16) __bf16 kl[64 * HD];       // K chunk  [key][c], 4 KB
    __shared__ __align__(16) __bf16 vl[HD * 64];       // V chunk  [c][key], 4 KB
    __shared__ __align__(16) __bf16 pbuf[4][32 * 64];  // per-wave P staging, 4 KB
    __bf16* pb = &pbuf[wave][0];

    // Two Q fragments (A layout 16x32), softmax scale folded in at QKV epilogue
    v16bf qa[2];
    const __bf16* qp = qbf + (bh * NSPAT + q0) * HD;
#pragma unroll
    for (int qg = 0; qg < 2; ++qg)
#pragma unroll
        for (int j = 0; j < 8; ++j) {
            int ka = half * 16 + 2 * j;
            qa[qg][2 * j]     = qp[(qg * 16 + l16) * HD + ka];
            qa[qg][2 * j + 1] = qp[(qg * 16 + l16) * HD + ka + 1];
        }

    float rowmax[2][8], lsum[2][8];
    v8f o00 = {}, o01 = {}, o10 = {}, o11 = {};
#pragma unroll
    for (int qg = 0; qg < 2; ++qg)
#pragma unroll
        for (int r = 0; r < 8; ++r) { rowmax[qg][r] = -3.0e38f; lsum[qg][r] = 0.f; }

    const __bf16* kp = kbf + bh * NSPAT * HD;
    const __bf16* vp = vbf + bh * HD * NSPAT;

    for (int kc = 0; kc < NSPAT; kc += 64) {
        // ---- cooperative staging of K / V^T chunks (16 B per thread x2) ----
        {
            const uint4* ksrc = (const uint4*)(kp + (size_t)kc * HD);  // 256 uint4
            uint4* kdst = (uint4*)kl;
            uint4* vdst = (uint4*)vl;
            for (int i = tid; i < 256; i += 128) {
                kdst[i] = ksrc[i];
                int c = i >> 3, k8 = i & 7;                            // 8 uint4 per c-row
                vdst[i] = *(const uint4*)(vp + (size_t)c * NSPAT + kc + k8 * 8);
            }
        }
        __builtin_prefetch(kp + (size_t)(kc + 64) * HD, 0, 1);
        __syncthreads();

#pragma unroll
        for (int qg = 0; qg < 2; ++qg) {
            // ---- S = Q K^T : four 16x16 tiles from LDS ----
            v8f s[4];
#pragma unroll
            for (int t = 0; t < 4; ++t) {
                v16bf kb;
#pragma unroll
                for (int j = 0; j < 8; ++j) {
                    int kk = half * 16 + 2 * j;                 // channel 0..31
                    kb[2 * j]     = kl[(t * 16 + l16) * HD + kk];
                    kb[2 * j + 1] = kl[(t * 16 + l16) * HD + kk + 1];
                }
                v8f z = {};
                s[t] = __builtin_amdgcn_wmma_f32_16x16x32_bf16(false, qa[qg], false, kb,
                                                               (short)0, z, false, false);
            }
            // ---- online softmax (rows live in 16-lane halves; xor 1,2,4,8) ----
#pragma unroll
            for (int r = 0; r < 8; ++r) {
                float m = fmaxf(fmaxf(s[0][r], s[1][r]), fmaxf(s[2][r], s[3][r]));
                m = fmaxf(m, __shfl_xor(m, 1));
                m = fmaxf(m, __shfl_xor(m, 2));
                m = fmaxf(m, __shfl_xor(m, 4));
                m = fmaxf(m, __shfl_xor(m, 8));
                float mnew = fmaxf(rowmax[qg][r], m);
                float corr = __expf(rowmax[qg][r] - mnew);
                rowmax[qg][r] = mnew;
                lsum[qg][r] *= corr;
                if (qg == 0) { o00[r] *= corr; o01[r] *= corr; }
                else         { o10[r] *= corr; o11[r] *= corr; }
                float rs = 0.f;
#pragma unroll
                for (int t = 0; t < 4; ++t) {
                    float p = __expf(s[t][r] - mnew);
                    s[t][r] = p;
                    rs += p;
                }
                rs += __shfl_xor(rs, 1);
                rs += __shfl_xor(rs, 2);
                rs += __shfl_xor(rs, 4);
                rs += __shfl_xor(rs, 8);
                lsum[qg][r] += rs;
            }
            // ---- P: C-layout -> wave-private LDS (DS in-order per wave) ----
#pragma unroll
            for (int t = 0; t < 4; ++t)
#pragma unroll
                for (int r = 0; r < 8; ++r) {
                    int row = qg * 16 + r + half * 8;
                    pb[row * 64 + t * 16 + l16] = (__bf16)s[t][r];
                }
            // ---- O += P V : 2 m-halves x 2 c-tiles ----
#pragma unroll
            for (int mh = 0; mh < 2; ++mh) {
                v16bf pa;
#pragma unroll
                for (int j = 0; j < 8; ++j) {
                    int ka = half * 16 + 2 * j;
                    pa[2 * j]     = pb[(qg * 16 + l16) * 64 + mh * 32 + ka];
                    pa[2 * j + 1] = pb[(qg * 16 + l16) * 64 + mh * 32 + ka + 1];
                }
#pragma unroll
                for (int ct = 0; ct < 2; ++ct) {
                    v16bf vb;
                    int c = ct * 16 + l16;
#pragma unroll
                    for (int j = 0; j < 8; ++j) {
                        int kk = mh * 32 + half * 16 + 2 * j;   // key_local
                        vb[2 * j]     = vl[c * 64 + kk];
                        vb[2 * j + 1] = vl[c * 64 + kk + 1];
                    }
                    if (qg == 0) {
                        if (ct == 0) o00 = __builtin_amdgcn_wmma_f32_16x16x32_bf16(false, pa, false, vb, (short)0, o00, false, false);
                        else         o01 = __builtin_amdgcn_wmma_f32_16x16x32_bf16(false, pa, false, vb, (short)0, o01, false, false);
                    } else {
                        if (ct == 0) o10 = __builtin_amdgcn_wmma_f32_16x16x32_bf16(false, pa, false, vb, (short)0, o10, false, false);
                        else         o11 = __builtin_amdgcn_wmma_f32_16x16x32_bf16(false, pa, false, vb, (short)0, o11, false, false);
                    }
                }
            }
        }
        __syncthreads();   // protect kl/vl before next chunk's staging
    }
    // ---- finalize: divide by row sum, scatter to ao [B][C][N] (fp32) ----
#pragma unroll
    for (int qg = 0; qg < 2; ++qg)
#pragma unroll
        for (int r = 0; r < 8; ++r) {
            int row = q0 + qg * 16 + r + half * 8;
            float inv = 1.0f / lsum[qg][r];
            float v0 = (qg == 0) ? o00[r] : o10[r];
            float v1 = (qg == 0) ? o01[r] : o11[r];
            int c = l16;
            ao[((size_t)(b * CCH + h * HD + c)) * NSPAT + row]      = v0 * inv;
            ao[((size_t)(b * CCH + h * HD + 16 + c)) * NSPAT + row] = v1 * inv;
        }
}

// ---------------------------------------------------------------------------
// Kernel 5: output projection GEMM + bias + residual (fp32 out).
// ---------------------------------------------------------------------------
__global__ void __launch_bounds__(128) outproj_gemm(const float* __restrict__ wout,
                                                    const float* __restrict__ ao,
                                                    const float* __restrict__ bout,
                                                    const float* __restrict__ x,
                                                    float* __restrict__ out) {
    __shared__ __align__(16) __bf16 lds_a[16 * 128];
    __shared__ __align__(16) __bf16 lds_b[64 * 128];
    int mt = blockIdx.x;                  // 0..7
    int n0 = blockIdx.y * 64;
    int b  = blockIdx.z;
    int tid = threadIdx.x;

    for (int i = tid; i < 16 * 128; i += 128) {
        int r = i >> 7, c0 = i & 127;
        lds_a[i] = (__bf16)wout[(mt * 16 + r) * 128 + c0];
    }
    for (int i = tid; i < 64 * 128; i += 128) {
        int c0 = i >> 6, j = i & 63;
        lds_b[j * 128 + c0] = (__bf16)ao[((size_t)(b * CCH + c0)) * NSPAT + n0 + j];
    }
    __syncthreads();

    int lane = tid & 31, wave = tid >> 5;
    int half = lane >> 4, l16 = lane & 15;
    v8f acc = {};
#pragma unroll
    for (int ks = 0; ks < 4; ++ks) {
        int kk0 = ks * 32;
        v16bf a, bb;
#pragma unroll
        for (int j = 0; j < 8; ++j) {
            int ka = half * 16 + 2 * j;
            a[2 * j]     = lds_a[l16 * 128 + kk0 + ka];
            a[2 * j + 1] = lds_a[l16 * 128 + kk0 + ka + 1];
            int kb = half * 16 + 2 * j;
            bb[2 * j]     = lds_b[(wave * 16 + l16) * 128 + kk0 + kb];
            bb[2 * j + 1] = lds_b[(wave * 16 + l16) * 128 + kk0 + kb + 1];
        }
        acc = __builtin_amdgcn_wmma_f32_16x16x32_bf16(false, a, false, bb,
                                                      (short)0, acc, false, false);
    }
#pragma unroll
    for (int r = 0; r < 8; ++r) {
        int o = mt * 16 + r + half * 8;
        int n = n0 + wave * 16 + l16;
        size_t idx = ((size_t)(b * CCH + o)) * NSPAT + n;
        out[idx] = acc[r] + bout[o] + x[idx];
    }
}

// ---------------------------------------------------------------------------
extern "C" void kernel_launch(void* const* d_in, const int* in_sizes, int n_in,
                              void* d_out, int out_size, void* d_ws, size_t ws_size,
                              hipStream_t stream) {
    (void)in_sizes; (void)n_in; (void)out_size; (void)ws_size;
    const float* x     = (const float*)d_in[0];
    const float* gamma = (const float*)d_in[1];
    const float* beta  = (const float*)d_in[2];
    const float* wqkv  = (const float*)d_in[3];
    const float* wout  = (const float*)d_in[4];
    const float* bout  = (const float*)d_in[5];
    float* out = (float*)d_out;

    char* ws = (char*)d_ws;
    const size_t MB = 1024u * 1024u;
    float*  stats  = (float*)ws;                         // 256 B (pad to 1 KB)
    __bf16* normbf = (__bf16*)(ws + 1024);               // 2 MB  [B][C][N]
    __bf16* qbf    = (__bf16*)(ws + 1024 + 2 * MB);      // 2 MB  [B*NH][N][HD]
    __bf16* kbf    = (__bf16*)(ws + 1024 + 4 * MB);      // 2 MB  [B*NH][N][HD]
    __bf16* vbf    = (__bf16*)(ws + 1024 + 6 * MB);      // 2 MB  [B*NH][HD][N]
    float*  ao     = (float*)(ws + 1024 + 8 * MB);       // 4 MB  [B][C][N]

    gn_stats<<<BATCH * NGRP, 256, 0, stream>>>(x, stats);
    gn_apply<<<(BATCH * CCH * NSPAT) / 256, 256, 0, stream>>>(x, gamma, beta, stats, normbf);
    qkv_gemm<<<dim3(24, NSPAT / 64, BATCH), 128, 0, stream>>>(wqkv, normbf, qbf, kbf, vbf);
    flash_attn<<<dim3(NSPAT / 128, NH, BATCH), 128, 0, stream>>>(qbf, kbf, vbf, ao);
    outproj_gemm<<<dim3(8, NSPAT / 64, BATCH), 128, 0, stream>>>(wout, ao, bout, x, out);
}
